// MultiHeadAttentionLayer_17506286698747
// MI455X (gfx1250) — compile-verified, matrix operand order checked by reading, not run on hardware
//
#include <hip/hip_runtime.h>
#include <hip/hip_bf16.h>
#include <cstdint>

typedef __attribute__((ext_vector_type(16))) _Float16 v16h;
typedef __attribute__((ext_vector_type(8)))  float    v8f;

#define N_NODES   50000
#define N_EDGES   800000
#define IN_DIM    128
#define P_DIM     128     // NUM_HEADS * OUT_DIM
#define NUM_HEADS 8
#define OUT_DIM   16
#define LDS_STRIDE 136    // 128 + 8 halves pad: 272B rows, 16B-aligned, bank-rotated

// ---- WMMA fragment loader (ISA 7.12.2 layout, wave32) ----------------------
// Works for both A (row-major activations, row = M) and B (col-major weights,
// row = N column) when the LDS tile is stored [16 rows][LDS_STRIDE] halves:
//   lanes 0-15 : row = lane,    K = k0+{0..7}  and k0+{16..23}
//   lanes 16-31: row = lane-16, K = k0+{8..15} and k0+{24..31}
// Each half is an aligned contiguous 16B run -> 2x ds_load_b128 per fragment.
__device__ inline v16h load_frag(const _Float16* __restrict__ tile, int lane, int k0) {
  const int hsel = lane >> 4;
  const int row  = lane & 15;
  const _Float16* p = tile + row * LDS_STRIDE + k0 + hsel * 8;
  v16h f;
#pragma unroll
  for (int j = 0; j < 8; ++j) f[j] = p[j];
#pragma unroll
  for (int j = 0; j < 8; ++j) f[8 + j] = p[16 + j];
  return f;
}

// B fragment from transposed weight LDS sWt[col][k] with stride LDS_STRIDE.
__device__ inline v16h load_frag_bt(const _Float16* __restrict__ wt, int lane,
                                    int k0, int colbase) {
  // rows of the "tile" are output columns colbase..colbase+15
  return load_frag(wt + colbase * LDS_STRIDE, lane, k0);
}

// ---- Kernel 1: node projections  out = h @ W + bias ------------------------
__global__ __launch_bounds__(256) void qkv_gemm_kernel(
    const float* __restrict__ h, const float* __restrict__ W,
    const float* __restrict__ bias, float* __restrict__ out) {
  __shared__ __align__(16) _Float16 sWt[P_DIM * LDS_STRIDE];  // transposed, 34 KB
  __shared__ __align__(16) _Float16 sA[64 * LDS_STRIDE];      // 17 KB
  __shared__ float sBias[P_DIM];
  const int tid = threadIdx.x;
  // W is [k][col] row-major; store transposed sWt[col][k] (coalesced global read)
  for (int i = tid; i < IN_DIM * P_DIM; i += 256) {
    int k = i >> 7, col = i & 127;
    sWt[col * LDS_STRIDE + k] = (_Float16)W[i];
  }
  if (tid < P_DIM) sBias[tid] = bias[tid];
  const int row0 = blockIdx.x * 64;
  for (int i = tid; i < 64 * IN_DIM; i += 256) {
    int r = i >> 7, cc = i & 127;
    int gr = row0 + r;
    sA[r * LDS_STRIDE + cc] =
        (_Float16)(gr < N_NODES ? h[(size_t)gr * IN_DIM + cc] : 0.0f);
  }
  __syncthreads();

  const int wave = tid >> 5, lane = tid & 31;
  const int coll = lane & 15;
  const int hsel = lane >> 4;
  const int col  = (wave << 4) + coll;
  // B fragments invariant across row sub-tiles: hoist (8x ds_load_b128 total)
  v16h b0 = load_frag_bt(sWt, lane,  0, wave << 4);
  v16h b1 = load_frag_bt(sWt, lane, 32, wave << 4);
  v16h b2 = load_frag_bt(sWt, lane, 64, wave << 4);
  v16h b3 = load_frag_bt(sWt, lane, 96, wave << 4);
  const float bval = sBias[col];
  const bool  full = (row0 + 64 <= N_NODES);   // uniform branch

#pragma unroll
  for (int sub = 0; sub < 4; ++sub) {
    const _Float16* tile = sA + sub * 16 * LDS_STRIDE;
    v8f c;
#pragma unroll
    for (int r = 0; r < 8; ++r) c[r] = bval;   // fold bias into accumulator
    v16h a;
    a = load_frag(tile, lane, 0);
    c = __builtin_amdgcn_wmma_f32_16x16x32_f16(false, a, false, b0, (short)0, c, false, false);
    a = load_frag(tile, lane, 32);
    c = __builtin_amdgcn_wmma_f32_16x16x32_f16(false, a, false, b1, (short)0, c, false, false);
    a = load_frag(tile, lane, 64);
    c = __builtin_amdgcn_wmma_f32_16x16x32_f16(false, a, false, b2, (short)0, c, false, false);
    a = load_frag(tile, lane, 96);
    c = __builtin_amdgcn_wmma_f32_16x16x32_f16(false, a, false, b3, (short)0, c, false, false);
    // D layout: VGPR r -> row (sub*16 + r + 8*hsel), col
    const int rloc = sub * 16 + (hsel << 3);
    float* op = out + (size_t)(row0 + rloc) * P_DIM + col;
    if (full) {
#pragma unroll
      for (int r = 0; r < 8; ++r) op[(size_t)r * P_DIM] = c[r];
    } else {
#pragma unroll
      for (int r = 0; r < 8; ++r)
        if (row0 + rloc + r < N_NODES) op[(size_t)r * P_DIM] = c[r];
    }
  }
}

// ---- Kernel 2: fused edge GEMM + scoring + clamp-exp + atomic scatter ------
// 64 edges/block; wave w == head w (OUT_DIM==16 => one N-tile per head).
__global__ __launch_bounds__(256) void edge_attn_kernel(
    const float* __restrict__ e, const float* __restrict__ We,
    const float* __restrict__ be, const int* __restrict__ src,
    const int* __restrict__ dst, const float* __restrict__ Q,
    const float* __restrict__ K, const float* __restrict__ V,
    float* __restrict__ score_out, float* __restrict__ wV,
    float* __restrict__ z) {
  __shared__ __align__(16) _Float16 sWt[P_DIM * LDS_STRIDE];  // 34 KB
  __shared__ __align__(16) _Float16 sE[64 * LDS_STRIDE];      // 17 KB
  __shared__ float sBias[P_DIM];
  __shared__ int   sSrc[64], sDst[64];
  __shared__ float sS[NUM_HEADS][16];
  const int tid = threadIdx.x;
  const int e0  = blockIdx.x * 64;            // N_EDGES % 64 == 0
  for (int i = tid; i < IN_DIM * P_DIM; i += 256) {
    int k = i >> 7, col = i & 127;
    sWt[col * LDS_STRIDE + k] = (_Float16)We[i];
  }
  if (tid < P_DIM) sBias[tid] = be[tid];
  for (int i = tid; i < 64 * IN_DIM; i += 256) {
    int r = i >> 7, cc = i & 127;
    sE[r * LDS_STRIDE + cc] = (_Float16)e[(size_t)(e0 + r) * IN_DIM + cc];
  }
  if (tid < 64) { sSrc[tid] = src[e0 + tid]; sDst[tid] = dst[e0 + tid]; }
  __syncthreads();

  const int wave = tid >> 5, lane = tid & 31;  // wave == head index
  const int coll = lane & 15;
  const int hsel = lane >> 4;
  const int colg = (wave << 4) + coll;         // global feature column
  v16h b0 = load_frag_bt(sWt, lane,  0, wave << 4);
  v16h b1 = load_frag_bt(sWt, lane, 32, wave << 4);
  v16h b2 = load_frag_bt(sWt, lane, 64, wave << 4);
  v16h b3 = load_frag_bt(sWt, lane, 96, wave << 4);
  const float biasv = sBias[colg];

  for (int sub = 0; sub < 4; ++sub) {
    const _Float16* tile = sE + sub * 16 * LDS_STRIDE;
    v8f c;
#pragma unroll
    for (int r = 0; r < 8; ++r) c[r] = biasv;  // bias folded into accumulator
    v16h a;
    a = load_frag(tile, lane, 0);
    c = __builtin_amdgcn_wmma_f32_16x16x32_f16(false, a, false, b0, (short)0, c, false, false);
    a = load_frag(tile, lane, 32);
    c = __builtin_amdgcn_wmma_f32_16x16x32_f16(false, a, false, b1, (short)0, c, false, false);
    a = load_frag(tile, lane, 64);
    c = __builtin_amdgcn_wmma_f32_16x16x32_f16(false, a, false, b2, (short)0, c, false, false);
    a = load_frag(tile, lane, 96);
    c = __builtin_amdgcn_wmma_f32_16x16x32_f16(false, a, false, b3, (short)0, c, false, false);

    // score + per-head row reduction (lanes 0-15 / 16-31 each hold one row)
    const int rloc = sub * 16 + (hsel << 3);               // first row this half
    float* sp = score_out + (size_t)(e0 + rloc) * P_DIM + colg;
#pragma unroll
    for (int r = 0; r < 8; ++r) {
      const int le = rloc + r;
      const int si = sSrc[le], di = sDst[le];
      const float sc = K[(size_t)si * P_DIM + colg] *
                       Q[(size_t)di * P_DIM + colg] * 0.25f * c[r];
      sp[(size_t)r * P_DIM] = sc;
      float t = sc;                            // sum over OUT_DIM = 16 lanes
#pragma unroll
      for (int m = 1; m < 16; m <<= 1) t += __shfl_xor(t, m, 16);
      t = __expf(fminf(fmaxf(t, -5.0f), 5.0f));
      if (coll == 0) sS[wave][(hsel << 3) + r] = t;
    }
    __syncthreads();
    // scatter: wV[dst] += V[src]*s ; z[dst] += s
#pragma unroll
    for (int r = 0; r < 8; ++r) {
      const int lrow = (hsel << 3) + r;
      const int le   = sub * 16 + lrow;
      const int si = sSrc[le], di = sDst[le];
      const float sv  = sS[wave][lrow];
      const float val = V[(size_t)si * P_DIM + colg] * sv;
      atomicAdd(&wV[(size_t)di * P_DIM + colg], val);
      if (coll == 0) atomicAdd(&z[(size_t)di * NUM_HEADS + wave], sv);
    }
    __syncthreads();
  }
}

// ---- helpers ---------------------------------------------------------------
__global__ void zero_kernel(float4* __restrict__ p, int n4) {
  int i = blockIdx.x * blockDim.x + threadIdx.x;
  if (i < n4) p[i] = make_float4(0.f, 0.f, 0.f, 0.f);
}

__global__ void finalize_kernel(const float4* __restrict__ wV,
                                const float* __restrict__ z,
                                float4* __restrict__ out) {
  int i = blockIdx.x * blockDim.x + threadIdx.x;   // over NP/4 float4s
  if (i < N_NODES * P_DIM / 4) {
    int idx = i << 2;
    int n   = idx >> 7;
    int hd  = (idx >> 4) & 7;                      // 4 dims share one head
    float inv = 1.0f / (z[n * NUM_HEADS + hd] + 1e-6f);
    float4 v = wV[i];
    out[i] = make_float4(v.x * inv, v.y * inv, v.z * inv, v.w * inv);
  }
}

// ---- launch ----------------------------------------------------------------
extern "C" void kernel_launch(void* const* d_in, const int* in_sizes, int n_in,
                              void* d_out, int out_size, void* d_ws, size_t ws_size,
                              hipStream_t stream) {
  const float* h  = (const float*)d_in[0];
  const float* e  = (const float*)d_in[1];
  const int*   src = (const int*)d_in[2];
  const int*   dst = (const int*)d_in[3];
  const float* Wq = (const float*)d_in[4];
  const float* bq = (const float*)d_in[5];
  const float* Wk = (const float*)d_in[6];
  const float* bk = (const float*)d_in[7];
  const float* We = (const float*)d_in[8];
  const float* be = (const float*)d_in[9];
  const float* Wv = (const float*)d_in[10];
  const float* bv = (const float*)d_in[11];

  const size_t NP = (size_t)N_NODES * P_DIM;  // 6.4M floats
  float* ws  = (float*)d_ws;                  // needs ~104 MB workspace
  float* Q   = ws;
  float* K   = ws + NP;
  float* V   = ws + 2 * NP;
  float* wV  = ws + 3 * NP;
  float* z   = ws + 4 * NP;                   // N_NODES * NUM_HEADS floats

  float* out_h     = (float*)d_out;           // [N, H, D] flat
  float* out_score = out_h + NP;              // [E, H, D] flat

  const int nz4 = (int)((NP + (size_t)N_NODES * NUM_HEADS) / 4);  // both 16B-multiples
  zero_kernel<<<(nz4 + 255) / 256, 256, 0, stream>>>((float4*)wV, nz4);

  const int nblk = (N_NODES + 63) / 64;
  qkv_gemm_kernel<<<nblk, 256, 0, stream>>>(h, Wq, bq, Q);
  qkv_gemm_kernel<<<nblk, 256, 0, stream>>>(h, Wk, bk, K);
  qkv_gemm_kernel<<<nblk, 256, 0, stream>>>(h, Wv, bv, V);

  edge_attn_kernel<<<N_EDGES / 64, 256, 0, stream>>>(
      e, We, be, src, dst, Q, K, V, out_score, wV, z);

  const int nf4 = (int)(NP / 4);
  finalize_kernel<<<(nf4 + 255) / 256, 256, 0, stream>>>(
      (const float4*)wV, z, (float4*)out_h);
}
